// TokenEncoder_37726992728815
// MI455X (gfx1250) — compile-verified, matrix operand order checked by reading, not run on hardware
//
#include <hip/hip_runtime.h>
#include <hip/hip_bf16.h>

typedef __attribute__((ext_vector_type(16))) _Float16 v16h;
typedef __attribute__((ext_vector_type(8)))  _Float16 v8h;
typedef __attribute__((ext_vector_type(8)))  float    v8f;

#define E_    1024
#define H_    16
#define DH_   64
#define DC_   256
#define DC1_  512
#define DR_   32
#define NCLS_ 201
#define B_    8
#define S_    1024
#define SEQ   1025          // S + cls
#define MR    (B_ * SEQ)    // 8200 rows
#define MP    8208          // padded to multiple of 16 (513 tiles)
#define N1    288           // DC + DR   (c_kv | k_rot)
#define N2    2048          // 2*DE      (k | v)
#define EPS_  1.1920929e-07f

// ---------------------------------------------------------------- workspace map
static constexpr size_t A1_OFF   = 0;                                  // MP*E f16
static constexpr size_t W1T_OFF  = A1_OFF   + (size_t)MP * E_ * 2;     // N1*E f16
static constexpr size_t B1_OFF   = W1T_OFF  + (size_t)N1 * E_ * 2;     // N1 f32 (pad 4096)
static constexpr size_t W2T_OFF  = B1_OFF   + 4096;                    // N2*DC f16
static constexpr size_t B2_OFF   = W2T_OFF  + (size_t)N2 * DC_ * 2;    // N2 f32
static constexpr size_t C1_OFF   = B2_OFF   + (size_t)N2 * 4;          // MP*N1 f32
static constexpr size_t A2_OFF   = C1_OFF   + (size_t)MP * N1 * 4;     // MP*DC f16
static constexpr size_t KROT_OFF = A2_OFF   + (size_t)MP * DC_ * 2;    // MP*DR f32
static constexpr size_t KV_OFF   = KROT_OFF + (size_t)MP * DR_ * 4;    // MP*N2 f32
static constexpr size_t CQ_OFF   = KV_OFF   + (size_t)MP * N2 * 4;     // B*DC1 f32
static constexpr size_t Q_OFF    = CQ_OFF   + (size_t)B_ * DC1_ * 4;   // B*E f32
static constexpr size_t QR_OFF   = Q_OFF    + (size_t)B_ * E_ * 4;     // B*DC1 f32
static constexpr size_t ATT_OFF  = QR_OFF   + (size_t)B_ * DC1_ * 4;   // B*E f32

// ---------------------------------------------------------------- small prep kernels
__global__ void transpose_w_f16(const float* __restrict__ W, _Float16* __restrict__ Wt,
                                int K, int N) {
  int idx = blockIdx.x * blockDim.x + threadIdx.x;
  if (idx >= K * N) return;
  int k = idx / N, n = idx % N;
  Wt[(size_t)n * K + k] = (_Float16)W[idx];
}

__global__ void copy_f32(const float* __restrict__ src, float* __restrict__ dst, int n) {
  int i = blockIdx.x * blockDim.x + threadIdx.x;
  if (i < n) dst[i] = src[i];
}

// x_ext = [cls; x]; A = f16(rmsnorm(x_ext) * rms_w), rows padded with zeros to MP
__global__ __launch_bounds__(256)
void prep_a_rmsnorm(const float* __restrict__ x, const float* __restrict__ cls,
                    const float* __restrict__ rmsw, _Float16* __restrict__ A) {
  __shared__ float red[256];
  int r = blockIdx.x, tid = threadIdx.x;
  _Float16* out = A + (size_t)r * E_;
  if (r >= MR) {
    for (int i = tid; i < E_; i += 256) out[i] = (_Float16)0.f;
    return;
  }
  int b = r / SEQ, s = r % SEQ;
  const float* src = (s == 0) ? cls : (x + ((size_t)b * S_ + (s - 1)) * E_);
  float v[4]; float ss = 0.f;
#pragma unroll
  for (int i = 0; i < 4; ++i) { v[i] = src[tid + 256 * i]; ss += v[i] * v[i]; }
  red[tid] = ss; __syncthreads();
  for (int off = 128; off > 0; off >>= 1) {
    if (tid < off) red[tid] += red[tid + off];
    __syncthreads();
  }
  float scale = __frsqrt_rn(red[0] / (float)E_ + EPS_);
#pragma unroll
  for (int i = 0; i < 4; ++i) {
    int j = tid + 256 * i;
    out[j] = (_Float16)(v[i] * scale * rmsw[j]);
  }
}

// ---------------------------------------------------------------- WMMA GEMM
// C[M x N] = A[M x K](f16) * Bt[N x K](f16)^T + bias[N]; one wave -> 16 x (16*NT) tile.
template<int NT>
__global__ __launch_bounds__(256)
void gemm_f16_wmma(const _Float16* __restrict__ A, const _Float16* __restrict__ Bt,
                   const float* __restrict__ bias, float* __restrict__ C,
                   int tilesM, int N, int K) {
  int wid  = (blockIdx.x * blockDim.x + threadIdx.x) >> 5;  // wave32
  int lane = threadIdx.x & 31;
  int nblks = N / (16 * NT);
  if (wid >= tilesM * nblks) return;                         // wave-uniform exit
  int mt = wid % tilesM;
  int nb = wid / tilesM;
  int m    = lane & 15;        // row within A tile / column within B tile
  int half = lane >> 4;        // K-half select per ISA 16-bit layout

  const _Float16* arow = A + (size_t)(mt * 16 + m) * K;
  const _Float16* brow[NT];
  v8f acc[NT];
#pragma unroll
  for (int t = 0; t < NT; ++t) {
    int n = nb * (16 * NT) + t * 16 + m;
    brow[t] = Bt + (size_t)n * K;
    float bv = bias[n];
    acc[t] = (v8f){bv, bv, bv, bv, bv, bv, bv, bv};
  }
  for (int k0 = 0; k0 < K; k0 += 32) {
    union { v8h h[2]; v16h v; } ua;
    ua.h[0] = *(const v8h*)(arow + k0 +      8 * half);
    ua.h[1] = *(const v8h*)(arow + k0 + 16 + 8 * half);
#pragma unroll
    for (int t = 0; t < NT; ++t) {
      union { v8h h[2]; v16h v; } ub;
      ub.h[0] = *(const v8h*)(brow[t] + k0 +      8 * half);
      ub.h[1] = *(const v8h*)(brow[t] + k0 + 16 + 8 * half);
      acc[t] = __builtin_amdgcn_wmma_f32_16x16x32_f16(
          false, ua.v, false, ub.v, (short)0, acc[t], false, false);
    }
  }
#pragma unroll
  for (int t = 0; t < NT; ++t) {
    int n = nb * (16 * NT) + t * 16 + m;
    float* crow = C + (size_t)(mt * 16 + 8 * half) * N + n;  // element j -> row M = j + 8*half
#pragma unroll
    for (int j = 0; j < 8; ++j) crow[(size_t)j * N] = acc[t][j];
  }
}

// ---------------------------------------------------------------- c_kv -> f16, RoPE(k_rot)
__global__ __launch_bounds__(288)
void ckv_epilogue(const float* __restrict__ C1, _Float16* __restrict__ A2,
                  float* __restrict__ krot) {
  int r = blockIdx.x, c = threadIdx.x;
  if (r >= MR) {
    if (c < DC_) A2[(size_t)r * DC_ + c] = (_Float16)0.f;
    return;
  }
  float val = C1[(size_t)r * N1 + c];
  if (c < DC_) {
    A2[(size_t)r * DC_ + c] = (_Float16)val;
  } else {
    int lane = c - DC_;                         // 0..31 : exactly one wave
    float other = __shfl_xor(val, 16, 32);      // rotate-half partner
    int i = lane & 15;
    float pos = (float)(r % SEQ);
    float inv = powf(10000.f, -(float)i / 16.f);
    float ang = pos * inv;
    float cs = cosf(ang), sn = sinf(ang);
    float res = (lane < 16) ? (val * cs - other * sn) : (val * cs + other * sn);
    krot[(size_t)r * DR_ + lane] = res;
  }
}

// ---------------------------------------------------------------- q path (row 0 per batch)
__global__ __launch_bounds__(256)
void q_stage1(const _Float16* __restrict__ A, const float* __restrict__ W_DQ,
              const float* __restrict__ b_DQ, float* __restrict__ cq) {
  __shared__ float h0[E_];
  int b = blockIdx.x, tid = threadIdx.x;
  const _Float16* src = A + (size_t)(b * SEQ) * E_;
  for (int i = tid; i < E_; i += 256) h0[i] = (float)src[i];
  __syncthreads();
  float a0 = b_DQ[tid], a1 = b_DQ[tid + 256];
  for (int i = 0; i < E_; ++i) {
    float h = h0[i];
    const float* w = W_DQ + (size_t)i * DC1_;
    a0 += h * w[tid];
    a1 += h * w[tid + 256];
  }
  cq[b * DC1_ + tid] = a0;
  cq[b * DC1_ + tid + 256] = a1;
}

__global__ __launch_bounds__(256)
void q_stage2(const float* __restrict__ cq, const float* __restrict__ W_UQ,
              const float* __restrict__ b_UQ, const float* __restrict__ W_QR,
              const float* __restrict__ b_QR, float* __restrict__ q,
              float* __restrict__ qr) {
  __shared__ float cs[DC1_];
  int b = blockIdx.x, tid = threadIdx.x;
  for (int i = tid; i < DC1_; i += 256) cs[i] = cq[b * DC1_ + i];
  __syncthreads();
  float qa[4] = {b_UQ[tid], b_UQ[tid + 256], b_UQ[tid + 512], b_UQ[tid + 768]};
  float ra[2] = {b_QR[tid], b_QR[tid + 256]};
  for (int i = 0; i < DC1_; ++i) {
    float c = cs[i];
    const float* wu = W_UQ + (size_t)i * E_;
    qa[0] += c * wu[tid];       qa[1] += c * wu[tid + 256];
    qa[2] += c * wu[tid + 512]; qa[3] += c * wu[tid + 768];
    const float* wr = W_QR + (size_t)i * DC1_;
    ra[0] += c * wr[tid];       ra[1] += c * wr[tid + 256];
  }
  q[b * E_ + tid] = qa[0];        q[b * E_ + tid + 256] = qa[1];
  q[b * E_ + tid + 512] = qa[2];  q[b * E_ + tid + 768] = qa[3];
  qr[b * DC1_ + tid] = ra[0];     qr[b * DC1_ + tid + 256] = ra[1];
  // pos=0 => RoPE on q_rot is identity
}

// ---------------------------------------------------------------- attention, query 0 only
__global__ __launch_bounds__(128)
void attn_q0(const float* __restrict__ KV, const float* __restrict__ krot,
             const float* __restrict__ q, const float* __restrict__ qr,
             const float* __restrict__ amask, float* __restrict__ att) {
  __shared__ float sc[SEQ];
  __shared__ float red[128];
  __shared__ float qv[DH_ + DR_];
  int b = blockIdx.x >> 4, h = blockIdx.x & 15;
  int tid = threadIdx.x;
  if (tid < DH_)             qv[tid] = q[b * E_ + h * DH_ + tid];
  else if (tid < DH_ + DR_)  qv[tid] = qr[b * DC1_ + h * DR_ + (tid - DH_)];
  __syncthreads();
  const float scale = __frsqrt_rn((float)(DH_ + DR_));
  float lmax = -3.4e38f;
  for (int k = tid; k < SEQ; k += 128) {
    size_t r = (size_t)(b * SEQ + k);
    const float* kp = KV + r * N2 + h * DH_;
    float acc = 0.f;
#pragma unroll
    for (int d = 0; d < DH_; d += 4)
      acc += qv[d] * kp[d] + qv[d + 1] * kp[d + 1] + qv[d + 2] * kp[d + 2] + qv[d + 3] * kp[d + 3];
    const float* rp = krot + r * DR_;
#pragma unroll
    for (int d = 0; d < DR_; ++d) acc += qv[DH_ + d] * rp[d];
    float am = (k == 0) ? 1.f : amask[b * S_ + k - 1];
    float s = acc * scale + (am - 1.f) * 1e9f;
    sc[k] = s;
    lmax = fmaxf(lmax, s);
  }
  red[tid] = lmax; __syncthreads();
  for (int off = 64; off > 0; off >>= 1) {
    if (tid < off) red[tid] = fmaxf(red[tid], red[tid + off]);
    __syncthreads();
  }
  float gmax = red[0]; __syncthreads();
  float lsum = 0.f;
  for (int k = tid; k < SEQ; k += 128) {
    float e = __expf(sc[k] - gmax);
    sc[k] = e; lsum += e;
  }
  red[tid] = lsum; __syncthreads();
  for (int off = 64; off > 0; off >>= 1) {
    if (tid < off) red[tid] += red[tid + off];
    __syncthreads();
  }
  float inv = 1.f / red[0];
  if (tid < DH_) {
    float acc = 0.f;
    for (int k = 0; k < SEQ; ++k)
      acc += sc[k] * KV[(size_t)(b * SEQ + k) * N2 + E_ + h * DH_ + tid];
    att[b * E_ + h * DH_ + tid] = acc * inv;
  }
}

// ---------------------------------------------------------------- fused head (8 rows)
__global__ __launch_bounds__(256)
void head_final(const float* __restrict__ att, const float* __restrict__ W_O,
                const float* __restrict__ b_O, const float* __restrict__ cls,
                const float* __restrict__ rmsw, const float* __restrict__ W_lin,
                const float* __restrict__ b_lin, const float* __restrict__ W_cls,
                const float* __restrict__ b_cls, float* __restrict__ out, int ncout) {
  __shared__ float sa[E_];    // att row, later h3
  __shared__ float sh0[E_];   // residual-1 output
  __shared__ float sh2[E_];   // rmsnorm-2 output
  __shared__ float red[256];
  int b = blockIdx.x, tid = threadIdx.x;
  for (int i = tid; i < E_; i += 256) sa[i] = att[b * E_ + i];
  __syncthreads();
  float o[4] = {0, 0, 0, 0};
  for (int i = 0; i < E_; ++i) {
    float a = sa[i];
    const float* w = W_O + (size_t)i * E_;
    o[0] += a * w[tid];       o[1] += a * w[tid + 256];
    o[2] += a * w[tid + 512]; o[3] += a * w[tid + 768];
  }
  float ss = 0.f;
#pragma unroll
  for (int t = 0; t < 4; ++t) {
    int j = tid + 256 * t;
    float h0 = o[t] + b_O[j] + cls[j];   // residual: x_ext row 0 == cls token
    sh0[j] = h0;
    ss += h0 * h0;
  }
  red[tid] = ss; __syncthreads();
  for (int off = 128; off > 0; off >>= 1) {
    if (tid < off) red[tid] += red[tid + off];
    __syncthreads();
  }
  float scale = __frsqrt_rn(red[0] / (float)E_ + EPS_);
#pragma unroll
  for (int t = 0; t < 4; ++t) {
    int j = tid + 256 * t;
    sh2[j] = sh0[j] * scale * rmsw[j];
  }
  __syncthreads();
  float l[4] = {0, 0, 0, 0};
  for (int i = 0; i < E_; ++i) {
    float hh = sh2[i];
    const float* w = W_lin + (size_t)i * E_;
    l[0] += hh * w[tid];       l[1] += hh * w[tid + 256];
    l[2] += hh * w[tid + 512]; l[3] += hh * w[tid + 768];
  }
  __syncthreads();
#pragma unroll
  for (int t = 0; t < 4; ++t) {
    int j = tid + 256 * t;
    float tv = l[t] + b_lin[j];
    float sg = 1.f / (1.f + __expf(-tv));
    sa[j] = tv * sg + sh0[j];            // silu + residual-2
  }
  __syncthreads();
  for (int c = tid; c < ncout; c += 256) {
    float acc = b_cls[c];
    for (int i = 0; i < E_; ++i) acc += sa[i] * W_cls[(size_t)i * NCLS_ + c];
    out[b * ncout + c] = acc;
  }
}

// ---------------------------------------------------------------- launch
extern "C" void kernel_launch(void* const* d_in, const int* in_sizes, int n_in,
                              void* d_out, int out_size, void* d_ws, size_t ws_size,
                              hipStream_t stream) {
  const float* x     = (const float*)d_in[0];
  // d_in[1] = return_NoF (handled via out_size)
  const float* amask = (const float*)d_in[2];
  const float* cls   = (const float*)d_in[3];
  const float* rmsw  = (const float*)d_in[4];
  const float* W_DQ  = (const float*)d_in[5];
  const float* b_DQ  = (const float*)d_in[6];
  const float* W_UQ  = (const float*)d_in[7];
  const float* b_UQ  = (const float*)d_in[8];
  const float* W_QR  = (const float*)d_in[9];
  const float* b_QR  = (const float*)d_in[10];
  const float* W_DKV = (const float*)d_in[11];
  const float* b_DKV = (const float*)d_in[12];
  const float* W_UK  = (const float*)d_in[13];
  const float* b_UK  = (const float*)d_in[14];
  const float* W_UV  = (const float*)d_in[15];
  const float* b_UV  = (const float*)d_in[16];
  const float* W_KR  = (const float*)d_in[17];
  const float* b_KR  = (const float*)d_in[18];
  const float* W_O   = (const float*)d_in[19];
  const float* b_O   = (const float*)d_in[20];
  const float* W_lin = (const float*)d_in[21];
  const float* b_lin = (const float*)d_in[22];
  const float* W_cls = (const float*)d_in[23];
  const float* b_cls = (const float*)d_in[24];

  char* ws = (char*)d_ws;
  _Float16* A1   = (_Float16*)(ws + A1_OFF);
  _Float16* W1t  = (_Float16*)(ws + W1T_OFF);
  float*    bias1= (float*)   (ws + B1_OFF);
  _Float16* W2t  = (_Float16*)(ws + W2T_OFF);
  float*    bias2= (float*)   (ws + B2_OFF);
  float*    C1   = (float*)   (ws + C1_OFF);
  _Float16* A2   = (_Float16*)(ws + A2_OFF);
  float*    krot = (float*)   (ws + KROT_OFF);
  float*    KV   = (float*)   (ws + KV_OFF);
  float*    cq   = (float*)   (ws + CQ_OFF);
  float*    qbuf = (float*)   (ws + Q_OFF);
  float*    qrbuf= (float*)   (ws + QR_OFF);
  float*    attb = (float*)   (ws + ATT_OFF);
  float*    outp = (float*)d_out;
  int ncout = out_size / B_;

  // weight prep: Bt = W^T in f16, bias concat
  transpose_w_f16<<<(E_ * DC_ + 255) / 256, 256, 0, stream>>>(W_DKV, W1t, E_, DC_);
  transpose_w_f16<<<(E_ * DR_ + 255) / 256, 256, 0, stream>>>(W_KR, W1t + (size_t)DC_ * E_, E_, DR_);
  transpose_w_f16<<<(DC_ * E_ + 255) / 256, 256, 0, stream>>>(W_UK, W2t, DC_, E_);
  transpose_w_f16<<<(DC_ * E_ + 255) / 256, 256, 0, stream>>>(W_UV, W2t + (size_t)E_ * DC_, DC_, E_);
  copy_f32<<<1, 256, 0, stream>>>(b_DKV, bias1, DC_);
  copy_f32<<<1, 32, 0, stream>>>(b_KR, bias1 + DC_, DR_);
  copy_f32<<<4, 256, 0, stream>>>(b_UK, bias2, E_);
  copy_f32<<<4, 256, 0, stream>>>(b_UV, bias2 + E_, E_);

  // activations: rmsnorm -> f16, padded
  prep_a_rmsnorm<<<MP, 256, 0, stream>>>(x, cls, rmsw, A1);

  // GEMM1: [MP x 1024] x [1024 x 288] -> c_kv | k_rot   (513 m-tiles, 9 n-blocks of 32)
  {
    int tilesM = MP / 16, nblks = N1 / 32;
    int waves = tilesM * nblks;
    gemm_f16_wmma<2><<<(waves + 7) / 8, 256, 0, stream>>>(A1, W1t, bias1, C1, tilesM, N1, E_);
  }
  ckv_epilogue<<<MP, 288, 0, stream>>>(C1, A2, krot);

  // GEMM2: [MP x 256] x [256 x 2048] -> k | v   (513 m-tiles, 32 n-blocks of 64)
  {
    int tilesM = MP / 16, nblks = N2 / 64;
    int waves = tilesM * nblks;
    gemm_f16_wmma<4><<<(waves + 7) / 8, 256, 0, stream>>>(A2, W2t, bias2, KV, tilesM, N2, DC_);
  }

  // q path (CLS row only)
  q_stage1<<<B_, 256, 0, stream>>>(A1, W_DQ, b_DQ, cq);
  q_stage2<<<B_, 256, 0, stream>>>(cq, W_UQ, b_UQ, W_QR, b_QR, qbuf, qrbuf);

  // attention for query position 0
  attn_q0<<<B_ * H_, 128, 0, stream>>>(KV, krot, qbuf, qrbuf, amask, attb);

  // fused head: W_O + residual + rmsnorm + silu(W_lin) + residual + W_cls
  head_final<<<B_, 256, 0, stream>>>(attb, W_O, b_O, cls, rmsw, W_lin, b_lin,
                                     W_cls, b_cls, outp, ncout);
}
